// ModelClass_43851616092233
// MI455X (gfx1250) — matrix operand order repaired and checked by reading, not compiled, not run on hardware
//
#include <hip/hip_runtime.h>

// ---------------------------------------------------------------------------
// Tree-growing GNN (F=G=64, BR=6, LAYERS=5, B=64) on gfx1250.
// All GEMMs on v_wmma_f32_16x16x32_bf16 (bf16 in, f32 accumulate).
// Each wave runs TWO independent WMMA accumulator chains (tile pairs) to fill
// the WMMA->WMMA RAW hazard window and halve A-fragment LDS traffic.
// Features stored f32 in d_out in the final concat layout [b][node][f].
// Weights converted+transposed to bf16 (N x K) in d_ws each launch.
// No FP atomics anywhere -> bit-deterministic given identical inputs.
// ---------------------------------------------------------------------------

typedef __attribute__((ext_vector_type(16))) __bf16 v16bf;
typedef __attribute__((ext_vector_type(8)))  float  v8f;

#define NTOT  9331
#define MAXBX 25      // max grid.x for k_pre = ceil(1555/64)

// RNE conversion: used only in one-time weight prep
__device__ __forceinline__ unsigned short f2bf_rne(float f) {
  union { float f; unsigned int u; } v; v.f = f;
  unsigned int r = v.u + 0x7FFFu + ((v.u >> 16) & 1u);
  return (unsigned short)(r >> 16);
}

// Truncating bf16 store: compiles to a plain d16_hi 16-bit store (zero ALU)
__device__ __forceinline__ void st_bf16(unsigned short* p, float x) {
  union { float f; unsigned int u; } v; v.f = x;
  *p = (unsigned short)(v.u >> 16);
}

union FragU { v16bf v; uint4 q[2]; };

// A fragment (16x32 bf16): lanes 0-15 rows M; halves = K {klo..+7, 16+klo..+7}
__device__ __forceinline__ v16bf load_a_frag(const unsigned short* A, int strideK,
                                             int m0, int k0) {
  const int lane = threadIdx.x & 31;
  const int m    = m0 + (lane & 15);
  const int klo  = (lane >> 4) << 3;               // 0 or 8
  FragU u;
  const unsigned short* p = A + m * strideK + k0 + klo;
  u.q[0] = *(const uint4*)p;
  u.q[1] = *(const uint4*)(p + 16);
  return u.v;
}

// B fragment (32x16 bf16) from weights stored N x K row-major:
// lane: N = lane&15, halves = K {k0+klo..+7, k0+16+klo..+7}, klo = lane<16 ? 0:8.
__device__ __forceinline__ v16bf load_b_frag(const unsigned short* Wt, int K,
                                             int n0, int k0) {
  const int lane = threadIdx.x & 31;
  const int n    = n0 + (lane & 15);
  const int kb   = k0 + ((lane >> 4) << 3);
  FragU u;
  const unsigned short* p = Wt + n * K + kb;
  u.q[0] = *(const uint4*)p;
  u.q[1] = *(const uint4*)(p + 16);
  return u.v;
}

struct C2 { v8f c0, c1; };

// Two 16x16 output tiles (same tm, columns tn0/tn1), interleaved WMMA chains.
template<int K>
__device__ __forceinline__ C2 wmma_tile2(const unsigned short* A,
                                         const unsigned short* Wt,
                                         int tm, int tn0, int tn1) {
  v8f z = {};
  C2 r; r.c0 = z; r.c1 = z;
#pragma unroll
  for (int k0 = 0; k0 < K; k0 += 32) {
    v16bf a  = load_a_frag(A, K, tm << 4, k0);
    v16bf b0 = load_b_frag(Wt, K, tn0 << 4, k0);
    v16bf b1 = load_b_frag(Wt, K, tn1 << 4, k0);
    r.c0 = __builtin_amdgcn_wmma_f32_16x16x32_bf16(false, a, false, b0, (short)0,
                                                   r.c0, false, false);
    r.c1 = __builtin_amdgcn_wmma_f32_16x16x32_bf16(false, a, false, b1, (short)0,
                                                   r.c1, false, false);
  }
  return r;
}

// Linear+ReLU on a 64-row LDS tile: Out(64xN bf16) = relu(A(64xK)@W + b).
// 256 threads = 8 waves; tile pairs divide evenly for N in {64,192,384}.
template<int K, int N>
__device__ void mlp_layer_t(const unsigned short* A, const unsigned short* Wt,
                            const float* bias, unsigned short* Out,
                            int ostride, int ocol0) {
  constexpr int NPC = N >> 5;     // column-tile pairs per row tile
  constexpr int NP  = 4 * NPC;    // total pairs
  for (int p = (int)(threadIdx.x >> 5); p < NP; p += 8) {
    const int tm = p / NPC, tj = p % NPC;
    C2 r = wmma_tile2<K>(A, Wt, tm, tj * 2, tj * 2 + 1);
    const int lane  = threadIdx.x & 31;
    const int col0  = (tj << 5) + (lane & 15);
    const float bs0 = bias[col0], bs1 = bias[col0 + 16];
    const int rbase = (tm << 4) + ((lane >> 4) << 3);
    unsigned short* o = Out + ocol0 + col0;
#pragma unroll
    for (int v = 0; v < 8; ++v) {
      st_bf16(o + (rbase + v) * ostride,      fmaxf(r.c0[v] + bs0, 0.f));
      st_bf16(o + (rbase + v) * ostride + 16, fmaxf(r.c1[v] + bs1, 0.f));
    }
  }
  __syncthreads();
}

// --------------------------- small utility kernels -------------------------

__global__ void k_prep_w(const float* __restrict__ W, unsigned short* __restrict__ Wt,
                         int din, int dout) {
  int i = blockIdx.x * blockDim.x + threadIdx.x;
  if (i < din * dout) {
    int k = i / dout, n = i % dout;
    Wt[n * din + k] = f2bf_rne(W[i]);   // transpose to N x K
  }
}

__global__ void k_copy_root(const float* __restrict__ rv, float* __restrict__ out) {
  out[(size_t)blockIdx.x * NTOT * 64 + threadIdx.x] = rv[blockIdx.x * 64 + threadIdx.x];
}

// ----------------------- GlobalDeepAggr: pre MLP + sum ---------------------
// grid (ceil(Ncur/64), B); per-block fixed-order column sums -> partial[b][bx]
__global__ void k_pre(const float* __restrict__ feats, int Ncur,
                      const unsigned short* w1, const float* b1,
                      const unsigned short* w2, const float* b2,
                      float* __restrict__ partial) {
  __shared__ unsigned short A[64 * 64];
  __shared__ unsigned short H1[64 * 64];
  __shared__ float H2F[64 * 64];
  const int b = blockIdx.y;
  const int row0 = blockIdx.x * 64;

  for (int i = threadIdx.x; i < 64 * 64; i += 256) {
    int r = i >> 6, f = i & 63;
    int node = row0 + r;
    float x = (node < Ncur) ? feats[((size_t)b * NTOT + node) * 64 + f] : 0.f;
    st_bf16(&A[i], x);
  }
  __syncthreads();

  mlp_layer_t<64, 64>(A, w1, b1, H1, 64, 0);

  {   // second layer: exactly one tile pair per wave, f32 out to LDS
    const int p  = (int)(threadIdx.x >> 5);
    const int tm = p >> 1, tj = p & 1;
    C2 r = wmma_tile2<64>(H1, w2, tm, tj * 2, tj * 2 + 1);
    const int lane  = threadIdx.x & 31;
    const int col0  = (tj << 5) + (lane & 15);
    const float bs0 = b2[col0], bs1 = b2[col0 + 16];
    const int rbase = (tm << 4) + ((lane >> 4) << 3);
#pragma unroll
    for (int v = 0; v < 8; ++v) {
      H2F[(rbase + v) * 64 + col0]      = fmaxf(r.c0[v] + bs0, 0.f);
      H2F[(rbase + v) * 64 + col0 + 16] = fmaxf(r.c1[v] + bs1, 0.f);
    }
  }
  __syncthreads();

  int nval = Ncur - row0; if (nval > 64) nval = 64;
  if (threadIdx.x < 64) {
    float s = 0.f;
    for (int r = 0; r < nval; ++r) s += H2F[r * 64 + threadIdx.x];   // fixed order
    partial[((size_t)b * MAXBX + blockIdx.x) * 64 + threadIdx.x] = s;
  }
}

// ----------------- post MLP: reduce partials, acc -> g ---------------------
__global__ void k_post(const float* __restrict__ partial, int nbx,
                       const unsigned short* w1, const float* b1,
                       const unsigned short* w2, const float* b2,
                       float* __restrict__ g) {
  __shared__ unsigned short A[64 * 64];
  __shared__ unsigned short H1[64 * 64];
  for (int i = threadIdx.x; i < 64 * 64; i += 256) {
    int b = i >> 6, f = i & 63;
    float s = 0.f;
    for (int x = 0; x < nbx; ++x) s += partial[((size_t)b * MAXBX + x) * 64 + f];
    st_bf16(&A[i], s);
  }
  __syncthreads();
  mlp_layer_t<64, 64>(A, w1, b1, H1, 64, 0);
  {
    const int p  = (int)(threadIdx.x >> 5);
    const int tm = p >> 1, tj = p & 1;
    C2 r = wmma_tile2<64>(H1, w2, tm, tj * 2, tj * 2 + 1);
    const int lane  = threadIdx.x & 31;
    const int col0  = (tj << 5) + (lane & 15);
    const float bs0 = b2[col0], bs1 = b2[col0 + 16];
    const int rbase = (tm << 4) + ((lane >> 4) << 3);
#pragma unroll
    for (int v = 0; v < 8; ++v) {
      g[(rbase + v) * 64 + col0]      = fmaxf(r.c0[v] + bs0, 0.f);
      g[(rbase + v) * 64 + col0 + 16] = fmaxf(r.c1[v] + bs1, 0.f);
    }
  }
}

// ---------------- NodeSpliter: [last,g] -> 384 -> 6 children ---------------
// dynamic LDS: A 64x128 + H1 64x384 + H2 64x384 (bf16) = 112 KB
__global__ void k_proj(const float* __restrict__ fin, int n,
                       const float* __restrict__ g,
                       const unsigned short* w1, const float* b1,
                       const unsigned short* w2, const float* b2,
                       const unsigned short* w3, const float* b3,
                       float* __restrict__ fchild) {
  extern __shared__ unsigned short L[];
  unsigned short* A  = L;                 // 64*128
  unsigned short* H1 = L + 64 * 128;      // 64*384
  unsigned short* H2 = H1 + 64 * 384;     // 64*384
  const int b = blockIdx.y;
  const int row0 = blockIdx.x * 64;

  for (int i = threadIdx.x; i < 64 * 128; i += 256) {
    int r = i >> 7, cx = i & 127;
    int node = row0 + r;
    float x;
    if (cx < 64) x = (node < n) ? fin[((size_t)b * NTOT + node) * 64 + cx] : 0.f;
    else         x = g[b * 64 + (cx - 64)];
    st_bf16(&A[i], x);
  }
  __syncthreads();

  mlp_layer_t<128, 384>(A,  w1, b1, H1, 384, 0);
  mlp_layer_t<384, 384>(H1, w2, b2, H2, 384, 0);

  for (int p = (int)(threadIdx.x >> 5); p < 48; p += 8) {
    const int tm = p / 12, tj = p % 12;
    C2 r = wmma_tile2<384>(H2, w3, tm, tj * 2, tj * 2 + 1);
    const int lane  = threadIdx.x & 31;
    const int col0  = (tj << 5) + (lane & 15);
    const int col1  = col0 + 16;
    const float bs0 = b3[col0], bs1 = b3[col1];
    const int rbase = (tm << 4) + ((lane >> 4) << 3);
#pragma unroll
    for (int v = 0; v < 8; ++v) {
      int node = row0 + rbase + v;
      if (node < n) {
        size_t base = (size_t)b * NTOT + (size_t)node * 6;
        fchild[(base + (col0 >> 6)) * 64 + (col0 & 63)] = fmaxf(r.c0[v] + bs0, 0.f);
        fchild[(base + (col1 >> 6)) * 64 + (col1 & 63)] = fmaxf(r.c1[v] + bs1, 0.f);
      }
    }
  }
}

// ------------- AncesterConv: msg(3) + upd(3) + 6-way mean, fused -----------
// dynamic LDS: BUF0 64x192 + BUF1 64x384 + CBUF 64x192 (bf16) + FOUT 64x64 f32
__global__ void k_anc(const float* __restrict__ pf,   // parent level (old)
                      const float* __restrict__ cf,   // child level (in-place out)
                      int nl, const float* __restrict__ g,
                      const unsigned short* mw1, const float* mb1,
                      const unsigned short* mw2, const float* mb2,
                      const unsigned short* mw3, const float* mb3,
                      const unsigned short* uw1, const float* ub1,
                      const unsigned short* uw2, const float* ub2,
                      const unsigned short* uw3, const float* ub3,
                      float* __restrict__ fout) {
  extern __shared__ unsigned short L[];
  unsigned short* BUF0 = L;                     // 64*192
  unsigned short* BUF1 = L + 64 * 192;          // 64*384
  unsigned short* CBUF = BUF1 + 64 * 384;       // 64*192
  float* FOUT = (float*)(CBUF + 64 * 192);      // 64*64
  const int b = blockIdx.y;
  const int row0 = blockIdx.x * 64;

  // msg input [parent, g] into BUF0 (stride 128)
  for (int i = threadIdx.x; i < 64 * 128; i += 256) {
    int r = i >> 7, cx = i & 127;
    int node = row0 + r;
    float x;
    if (cx < 64) x = (node < nl) ? pf[((size_t)b * NTOT + (node / 6)) * 64 + cx] : 0.f;
    else         x = g[b * 64 + (cx - 64)];
    st_bf16(&BUF0[i], x);
  }
  // upd concat buffer: cols 0..63 = child, 128..191 = g
  for (int i = threadIdx.x; i < 64 * 64; i += 256) {
    int r = i >> 6, f = i & 63;
    int node = row0 + r;
    float xc = (node < nl) ? cf[((size_t)b * NTOT + node) * 64 + f] : 0.f;
    st_bf16(&CBUF[r * 192 + f], xc);
    st_bf16(&CBUF[r * 192 + 128 + f], g[b * 64 + f]);
  }
  __syncthreads();

  mlp_layer_t<128, 64>(BUF0, mw1, mb1, BUF1, 64, 0);    // 128 -> 64
  mlp_layer_t< 64, 64>(BUF1, mw2, mb2, BUF0, 64, 0);    // 64  -> 64
  mlp_layer_t< 64, 64>(BUF0, mw3, mb3, CBUF, 192, 64);  // m into concat cols 64..127
  mlp_layer_t<192,192>(CBUF, uw1, ub1, BUF0, 192, 0);   // 192 -> 192
  mlp_layer_t<192,384>(BUF0, uw2, ub2, BUF1, 384, 0);   // 192 -> 384

  // final 384->384 with fused deterministic mean over BR=6 chunks:
  // 16 output groups (tm x 16-col group), 2 per wave; chunks accumulated in regs
  {
    const int lane = threadIdx.x & 31;
    for (int gi = (int)(threadIdx.x >> 5); gi < 16; gi += 8) {
      const int tm = gi >> 2, tc = gi & 3;
      v8f s = {};
#pragma unroll
      for (int j = 0; j < 6; j += 2) {
        const int tn0 = j * 4 + tc, tn1 = (j + 1) * 4 + tc;
        C2 r = wmma_tile2<384>(BUF1, uw3, tm, tn0, tn1);
        const float bs0 = ub3[(tn0 << 4) + (lane & 15)];
        const float bs1 = ub3[(tn1 << 4) + (lane & 15)];
#pragma unroll
        for (int v = 0; v < 8; ++v)
          s[v] += fmaxf(r.c0[v] + bs0, 0.f) + fmaxf(r.c1[v] + bs1, 0.f);
      }
      const int rbase = (tm << 4) + ((lane >> 4) << 3);
      const int col   = (tc << 4) + (lane & 15);
#pragma unroll
      for (int v = 0; v < 8; ++v)
        FOUT[(rbase + v) * 64 + col] = s[v] * (1.f / 6.f);
    }
  }
  __syncthreads();

  for (int i = threadIdx.x; i < 64 * 64; i += 256) {
    int r = i >> 6, f = i & 63;
    int node = row0 + r;
    if (node < nl) fout[((size_t)b * NTOT + node) * 64 + f] = FOUT[i];
  }
}

// ------------------------------- host side ---------------------------------

extern "C" void kernel_launch(void* const* d_in, const int* in_sizes, int n_in,
                              void* d_out, int out_size, void* d_ws, size_t ws_size,
                              hipStream_t stream) {
  (void)in_sizes; (void)n_in; (void)out_size; (void)ws_size;
  const float* rv = (const float*)d_in[0];
  // order: pre1,pre2,post1,post2,proj1..3,msg1..3,upd1..3
  static const int widx[13] = {1,3,5,7,9,11,13,15,17,19,21,23,25};
  static const int din_[13] = {64,64,64,64,128,384,384,128,64,64,192,192,384};
  static const int dout_[13]= {64,64,64,64,384,384,384,64,64,64,192,384,384};

  unsigned short* wt[13];
  const float* bias[13];
  size_t off = 0;
  for (int i = 0; i < 13; ++i) {
    wt[i] = (unsigned short*)((char*)d_ws + off);
    size_t e = (size_t)din_[i] * dout_[i];
    off += ((e * 2 + 255) / 256) * 256;
    bias[i] = (const float*)d_in[widx[i] + 1];
  }
  float* partial = (float*)((char*)d_ws + off); off += (size_t)64 * MAXBX * 64 * 4;
  float* gbuf    = (float*)((char*)d_ws + off); off += 64 * 64 * 4;
  float* out     = (float*)d_out;

  const int DLDS = 114688;  // 112 KB dynamic LDS for k_proj / k_anc
  hipFuncSetAttribute(reinterpret_cast<const void*>(k_proj),
                      hipFuncAttributeMaxDynamicSharedMemorySize, DLDS);
  hipFuncSetAttribute(reinterpret_cast<const void*>(k_anc),
                      hipFuncAttributeMaxDynamicSharedMemorySize, DLDS);

  for (int i = 0; i < 13; ++i) {
    int e = din_[i] * dout_[i];
    k_prep_w<<<dim3((e + 255) / 256), dim3(256), 0, stream>>>(
        (const float*)d_in[widx[i]], wt[i], din_[i], dout_[i]);
  }
  k_copy_root<<<dim3(64), dim3(64), 0, stream>>>(rv, out);

  static const int counts[6] = {1, 6, 36, 216, 1296, 7776};
  static const int bases[7]  = {0, 1, 7, 43, 259, 1555, 9331};

  for (int inx = 0; inx < 5; ++inx) {
    const int Ncur = bases[inx + 1];
    const int nbx  = (Ncur + 63) / 64;
    k_pre<<<dim3(nbx, 64), dim3(256), 0, stream>>>(
        out, Ncur, wt[0], bias[0], wt[1], bias[1], partial);
    k_post<<<dim3(1), dim3(256), 0, stream>>>(
        partial, nbx, wt[2], bias[2], wt[3], bias[3], gbuf);

    const int n = counts[inx];
    k_proj<<<dim3((n + 63) / 64, 64), dim3(256), DLDS, stream>>>(
        out + (size_t)bases[inx] * 64, n, gbuf,
        wt[4], bias[4], wt[5], bias[5], wt[6], bias[6],
        out + (size_t)bases[inx + 1] * 64);

    // descending levels: level l reads only old levels l and l-1
    for (int l = inx + 1; l >= 1; --l) {
      const int nl = counts[l];
      k_anc<<<dim3((nl + 63) / 64, 64), dim3(256), DLDS, stream>>>(
          out + (size_t)bases[l - 1] * 64, out + (size_t)bases[l] * 64, nl, gbuf,
          wt[7], bias[7], wt[8], bias[8], wt[9], bias[9],
          wt[10], bias[10], wt[11], bias[11], wt[12], bias[12],
          out + (size_t)bases[l] * 64);
    }
  }
}
